// HGNN_res_49246095016349
// MI455X (gfx1250) — compile-verified, matrix operand order checked by reading, not run on hardware
//
#include <hip/hip_runtime.h>
#include <hip/hip_bf16.h>

// ---------- types ----------
typedef __attribute__((ext_vector_type(16))) __bf16       v16bf;
typedef __attribute__((ext_vector_type(8)))  float        v8f;
typedef __attribute__((ext_vector_type(4)))  float        v4f;
typedef __attribute__((ext_vector_type(4)))  unsigned int v4u;

union Frag { v16bf v; v4u q[2]; };

#define MAXN 0.99999f   /* (1-EPS)/sqrt(c), c=1 */

// ---------- device helpers ----------
__device__ inline float wave_sum(float v) {
#pragma unroll
  for (int off = 16; off > 0; off >>= 1) v += __shfl_xor(v, off, 32);
  return v;
}
__device__ inline float dot4(v4f a, v4f b) {
  return a.x * b.x + a.y * b.y + a.z * b.z + a.w * b.w;
}
__device__ inline float artanh_f(float x) {
  x = fminf(fmaxf(x, -1.f + 1e-5f), 1.f - 1e-5f);
  return 0.5f * (log1pf(x) - log1pf(-x));
}
__device__ inline unsigned short f2bf(float f) {  // round-to-nearest-even
  unsigned u = __float_as_uint(f);
  unsigned r = u + 0x7FFFu + ((u >> 16) & 1u);
  return (unsigned short)(r >> 16);
}

// CDNA5 async global->LDS copy (ASYNCcnt-tracked, no VGPR staging).
// LDS operand is the wave-relative LDS byte address = low 32 bits of the
// generic pointer to a __shared__ object (flat-LDS aperture layout).
__device__ inline unsigned lds_addr32(const void* p) {
  return (unsigned)(unsigned long long)p;
}
__device__ inline void async_ld_b128(unsigned lds_off, const void* gaddr) {
  asm volatile("global_load_async_to_lds_b128 %0, %1, off"
               :: "v"(lds_off), "v"(gaddr) : "memory");
}
__device__ inline void wait_async0() {
  asm volatile("s_wait_asynccnt 0x0" ::: "memory");
}

// ---------- fp32 -> bf16 packed converter ----------
__global__ void k_f32_to_bf16_pk(const float* __restrict__ s,
                                 unsigned* __restrict__ d, long long npairs) {
  long long i = (long long)blockIdx.x * blockDim.x + threadIdx.x;
  long long st = (long long)gridDim.x * blockDim.x;
  for (; i < npairs; i += st) {
    float a = s[2 * i], b = s[2 * i + 1];
    d[i] = (unsigned)f2bf(a) | ((unsigned)f2bf(b) << 16);
  }
}

// ---------- WMMA bf16 GEMM: C[M,128] = A[M,K] * B^T-layout, f32 accumulate --
// A: [M][K] bf16 row-major.  B: [128][K] bf16 row-major ("Bt": output column n
// is row n of B).  Double-buffered async-to-LDS pipeline, 1 barrier/iter.
// Block: 256 threads = 8 waves. BM=32, BN=128, BK=32.
// Wave w: rows 16*(w>>2)..+15, cols 32*(w&3)..+31 (two 16x16 acc tiles).
__global__ __launch_bounds__(256) void k_gemm_bf16(
    const unsigned short* __restrict__ A, const unsigned short* __restrict__ B,
    float* __restrict__ C, int K) {
  __shared__ unsigned short lA[2][32 * 32];   // [buf][row][k]
  __shared__ unsigned short lB[2][128 * 32];  // [buf][n][k]
  const int tid = threadIdx.x;
  const int wave = tid >> 5, lane = tid & 31;
  const int laneLo = lane & 15, laneHi = lane >> 4;
  const int rowHalf = wave >> 2, colQ = wave & 3;
  const size_t m0 = (size_t)blockIdx.x * 32;
  const int ar = tid >> 2, asg = tid & 3;  // A-tile assignment (threads 0..127)

  auto issue = [&](int buf, int k0) {
    if (tid < 128) {  // A tile: 32 rows x 32 bf16, 16B per thread
      async_ld_b128(lds_addr32(&lA[buf][ar * 32 + asg * 8]),
                    A + (m0 + ar) * (size_t)K + k0 + asg * 8);
    }
#pragma unroll
    for (int i = 0; i < 2; i++) {  // B tile: 128 rows x 32 bf16
      int si = tid + i * 256;
      int r = si >> 2, sg = si & 3;
      async_ld_b128(lds_addr32(&lB[buf][r * 32 + sg * 8]),
                    B + (size_t)r * K + k0 + sg * 8);
    }
  };

  v8f acc0 = {0.f, 0.f, 0.f, 0.f, 0.f, 0.f, 0.f, 0.f};
  v8f acc1 = acc0;

  issue(0, 0);
  int cur = 0;
  for (int k0 = 0; k0 < K; k0 += 32) {
    wait_async0();      // this wave's tile-`cur` transfers are done
    __syncthreads();    // everyone's are done; prev-iter LDS readers are past
    if (k0 + 32 < K) issue(cur ^ 1, k0 + 32);  // overlap next tile with compute
    // fragments (ISA 16x32 bf16 layout): lanes0-15 K{0..7,16..23},
    // lanes16-31 K{8..15,24..31}; two ds_read_b128 per fragment.
    Frag a, b0, b1;
    {
      const v4u* p = (const v4u*)(&lA[cur][(rowHalf * 16 + laneLo) * 32]);
      a.q[0] = p[laneHi]; a.q[1] = p[2 + laneHi];
    }
    {
      const v4u* p = (const v4u*)(&lB[cur][(colQ * 32 + laneLo) * 32]);
      b0.q[0] = p[laneHi]; b0.q[1] = p[2 + laneHi];
    }
    {
      const v4u* p = (const v4u*)(&lB[cur][(colQ * 32 + 16 + laneLo) * 32]);
      b1.q[0] = p[laneHi]; b1.q[1] = p[2 + laneHi];
    }
    acc0 = __builtin_amdgcn_wmma_f32_16x16x32_bf16(false, a.v, false, b0.v,
                                                   (short)0, acc0, false, false);
    acc1 = __builtin_amdgcn_wmma_f32_16x16x32_bf16(false, a.v, false, b1.v,
                                                   (short)0, acc1, false, false);
    cur ^= 1;
  }
  // D layout: VGPR r -> row (laneHi*8 + r), col laneLo
  int colBase = colQ * 32 + laneLo;
  size_t rowBase = m0 + rowHalf * 16 + laneHi * 8;
#pragma unroll
  for (int r = 0; r < 8; r++) {
    C[(rowBase + r) * 128 + colBase] = acc0[r];
    C[(rowBase + r) * 128 + colBase + 16] = acc1[r];
  }
}

// ---------- h0 = proj(expmap0(x)) : one wave per 128-dim row ----------
__global__ __launch_bounds__(256) void k_init_h(const float* __restrict__ x,
                                                float* __restrict__ h, int N) {
  int wid = (int)((blockIdx.x * blockDim.x + threadIdx.x) >> 5);
  int lane = threadIdx.x & 31;
  if (wid >= N) return;
  size_t base = (size_t)wid * 128 + lane * 4;
  v4f v = *(const v4f*)(x + base);
  float nr = fmaxf(sqrtf(wave_sum(dot4(v, v))), 1e-15f);
  float es = tanhf(nr) / nr;
  float en = fmaxf(es * nr, 1e-15f);
  float s = es * ((en > MAXN) ? (MAXN / en) : 1.f);
  v4f o; o.x = s * v.x; o.y = s * v.y; o.z = s * v.z; o.w = s * v.w;
  *(v4f*)(h + base) = o;
}

// ---------- HypLinear tail: mobius_matvec scale + proj + hyperbolic bias +
// proj; writes h and t = logmap0(h) TRANSPOSED in bf16 (tbT[feature][node])
// so the adjacency GEMM's B operand is contiguous. ----------
__global__ __launch_bounds__(256) void k_hl_post(
    const float* __restrict__ mx, float* __restrict__ h,
    const float* __restrict__ bvec, unsigned short* __restrict__ tbT, int N) {
  int wid = (int)((blockIdx.x * blockDim.x + threadIdx.x) >> 5);
  int lane = threadIdx.x & 31;
  if (wid >= N) return;
  size_t base = (size_t)wid * 128 + lane * 4;
  v4f hx = *(const v4f*)(h + base);
  v4f mv = *(const v4f*)(mx + base);
  v4f bv = *(const v4f*)(bvec + lane * 4);
  float xn2 = wave_sum(dot4(hx, hx));
  float mn2 = wave_sum(dot4(mv, mv));
  float bn2 = wave_sum(dot4(bv, bv));
  float mb = wave_sum(dot4(mv, bv));
  float x_norm = fmaxf(sqrtf(xn2), 1e-15f);
  float mx_norm = fmaxf(sqrtf(mn2), 1e-15f);
  // mobius_matvec scale then proj  ->  mv_final = alpha * mx
  float s1 = tanhf(mx_norm / x_norm * artanh_f(x_norm)) / mx_norm;
  float mvn = fmaxf(s1 * mx_norm, 1e-15f);
  float alpha = s1 * ((mvn > MAXN) ? (MAXN / mvn) : 1.f);
  float x2 = alpha * alpha * mn2;
  // hyp_b = proj(expmap0(b))  ->  y = beta * b
  float b_norm = fmaxf(sqrtf(bn2), 1e-15f);
  float es = tanhf(b_norm) / b_norm;
  float ebn = fmaxf(es * b_norm, 1e-15f);
  float beta = es * ((ebn > MAXN) ? (MAXN / ebn) : 1.f);
  float y2 = beta * beta * bn2;
  float xy = alpha * beta * mb;
  // mobius_add(mv, y), c=1
  float Ac = 1.f + 2.f * xy + y2;
  float Bc = 1.f - x2;
  float den = fmaxf(1.f + 2.f * xy + x2 * y2, 1e-15f);
  float ca = Ac * alpha / den, cb = Bc * beta / den;
  float on2 = (Ac * Ac * x2 + 2.f * Ac * Bc * xy + Bc * Bc * y2) / (den * den);
  float on = fmaxf(sqrtf(fmaxf(on2, 0.f)), 1e-15f);
  float ops = (on > MAXN) ? (MAXN / on) : 1.f;
  ca *= ops; cb *= ops; on = fmaxf(on * ops, 1e-15f);
  v4f o;
  o.x = ca * mv.x + cb * bv.x; o.y = ca * mv.y + cb * bv.y;
  o.z = ca * mv.z + cb * bv.z; o.w = ca * mv.w + cb * bv.w;
  *(v4f*)(h + base) = o;
  // t = logmap0(h_new), stored transposed: tbT[d * N + node]
  float ls = artanh_f(on) / on;
  int d0 = lane * 4;
  tbT[(size_t)(d0 + 0) * N + wid] = f2bf(ls * o.x);
  tbT[(size_t)(d0 + 1) * N + wid] = f2bf(ls * o.y);
  tbT[(size_t)(d0 + 2) * N + wid] = f2bf(ls * o.z);
  tbT[(size_t)(d0 + 3) * N + wid] = f2bf(ls * o.w);
}

// ---------- HypAgg+HypAct: h = proj(expmap0(relu(logmap0(proj(expmap0(u))))))
// (+ optional Euclidean residual, faithful to reference) ----------
__global__ __launch_bounds__(256) void k_agg_act(
    const float* __restrict__ u, float* __restrict__ h,
    const float* __restrict__ res, int add_res, int N) {
  int wid = (int)((blockIdx.x * blockDim.x + threadIdx.x) >> 5);
  int lane = threadIdx.x & 31;
  if (wid >= N) return;
  size_t base = (size_t)wid * 128 + lane * 4;
  v4f uv = *(const v4f*)(u + base);
  float un = fmaxf(sqrtf(wave_sum(dot4(uv, uv))), 1e-15f);
  float e1 = tanhf(un) / un;
  float vn = fmaxf(e1 * un, 1e-15f);
  float p1 = (vn > MAXN) ? (MAXN / vn) : 1.f;
  float g = e1 * p1;
  vn = fmaxf(vn * p1, 1e-15f);
  float gw = g * (artanh_f(vn) / vn);  // logmap0 of projected point
  v4f w;
  w.x = fmaxf(gw * uv.x, 0.f); w.y = fmaxf(gw * uv.y, 0.f);
  w.z = fmaxf(gw * uv.z, 0.f); w.w = fmaxf(gw * uv.w, 0.f);
  float wn = fmaxf(sqrtf(wave_sum(dot4(w, w))), 1e-15f);
  float e2 = tanhf(wn) / wn;
  float en = fmaxf(e2 * wn, 1e-15f);
  float s = e2 * ((en > MAXN) ? (MAXN / en) : 1.f);
  v4f o; o.x = s * w.x; o.y = s * w.y; o.z = s * w.z; o.w = s * w.w;
  if (add_res) {
    v4f rv = *(const v4f*)(res + base);
    o.x += rv.x; o.y += rv.y; o.z += rv.z; o.w += rv.w;
  }
  *(v4f*)(h + base) = o;
}

// ---------- centroid squared norms ----------
__global__ void k_centnorm(const float* __restrict__ cent, float* __restrict__ cn2) {
  int wid = (int)((blockIdx.x * blockDim.x + threadIdx.x) >> 5);
  int lane = threadIdx.x & 31;
  if (wid >= 64) return;
  v4f v = *(const v4f*)(cent + (size_t)wid * 128 + lane * 4);
  float n2 = wave_sum(dot4(v, v));
  if (!lane) cn2[wid] = n2;
}

// ---------- poincare_dist(h[n], centroid[c]) via the 3 dot products ----------
__global__ void k_centdist(const float* __restrict__ h, const float* __restrict__ cent,
                           const float* __restrict__ cn2, float* __restrict__ dist) {
  __shared__ float sh[128];
  int n = blockIdx.x, t = threadIdx.x;  // 64 threads = 64 centroids
  sh[t] = h[(size_t)n * 128 + t];
  sh[t + 64] = h[(size_t)n * 128 + t + 64];
  __syncthreads();
  float h2 = 0.f;
#pragma unroll 4
  for (int i = 0; i < 128; i++) h2 += sh[i] * sh[i];
  const float* ce = cent + (size_t)t * 128;
  float ab = 0.f;
#pragma unroll 4
  for (int i = 0; i < 128; i++) ab += sh[i] * ce[i];
  ab = -ab;  // mobius_add(-x, y): <a,y> with a=-h
  float b2 = cn2[t];
  float A = 1.f + 2.f * ab + b2;
  float B = 1.f - h2;
  float den = fmaxf(1.f + 2.f * ab + h2 * b2, 1e-15f);
  float n2 = A * A * h2 + 2.f * A * B * ab + B * B * b2;
  float d = fmaxf(sqrtf(fmaxf(n2, 0.f)) / den, 1e-15f);
  dist[(size_t)n * 64 + t] = 2.f * artanh_f(d);
}

// ---------- readout = column means of dist ----------
__global__ void k_readout(const float* __restrict__ dist, float* __restrict__ ro) {
  __shared__ float red[256];
  int c = blockIdx.x, tid = threadIdx.x;
  float s = 0.f;
  for (int i = tid; i < 8192; i += 256) s += dist[(size_t)i * 64 + c];
  red[tid] = s; __syncthreads();
  for (int o = 128; o > 0; o >>= 1) {
    if (tid < o) red[tid] += red[tid + o];
    __syncthreads();
  }
  if (!tid) ro[c] = red[0] / 8192.f;
}

// ---------- h_r = logmap0(readout) (64 dims, one wave) ----------
__global__ void k_hr(const float* __restrict__ ro, float* __restrict__ out) {
  int lane = threadIdx.x;
  float a = ro[2 * lane], b = ro[2 * lane + 1];
  float nr = fmaxf(sqrtf(wave_sum(a * a + b * b)), 1e-15f);
  float ls = artanh_f(nr) / nr;
  out[2 * lane] = ls * a;
  out[2 * lane + 1] = ls * b;
}

// ---------- logits = softmax over length-1 axis == all ones ----------
__global__ void k_ones(float* __restrict__ out) {
  if (threadIdx.x < 7) out[threadIdx.x] = 1.0f;
}

// ---------- z = logmap0(h) @ W_mlp1^T + b ----------
__global__ __launch_bounds__(256) void k_node_logits(
    const float* __restrict__ h, const float* __restrict__ W,
    const float* __restrict__ b, float* __restrict__ z, int N) {
  int wid = (int)((blockIdx.x * blockDim.x + threadIdx.x) >> 5);
  int lane = threadIdx.x & 31;
  if (wid >= N) return;
  v4f hv = *(const v4f*)(h + (size_t)wid * 128 + lane * 4);
  float nr = fmaxf(sqrtf(wave_sum(dot4(hv, hv))), 1e-15f);
  float ls = artanh_f(nr) / nr;
  v4f h1; h1.x = ls * hv.x; h1.y = ls * hv.y; h1.z = ls * hv.z; h1.w = ls * hv.w;
#pragma unroll
  for (int j = 0; j < 7; j++) {
    v4f wv = *(const v4f*)(W + j * 128 + lane * 4);
    float d = wave_sum(dot4(h1, wv));
    if (!lane) z[(size_t)wid * 7 + j] = d + b[j];
  }
}

// ---------- softmax over axis 0 (8192 nodes) per class column ----------
__global__ void k_colsoftmax(const float* __restrict__ z, float* __restrict__ out, int N) {
  __shared__ float red[256];
  int c = blockIdx.x, tid = threadIdx.x;
  float m = -3.4e38f;
  for (int i = tid; i < N; i += 256) m = fmaxf(m, z[(size_t)i * 7 + c]);
  red[tid] = m; __syncthreads();
  for (int o = 128; o > 0; o >>= 1) {
    if (tid < o) red[tid] = fmaxf(red[tid], red[tid + o]);
    __syncthreads();
  }
  float M = red[0]; __syncthreads();
  float s = 0.f;
  for (int i = tid; i < N; i += 256) s += expf(z[(size_t)i * 7 + c] - M);
  red[tid] = s; __syncthreads();
  for (int o = 128; o > 0; o >>= 1) {
    if (tid < o) red[tid] += red[tid + o];
    __syncthreads();
  }
  float S = red[0];
  for (int i = tid; i < N; i += 256)
    out[(size_t)i * 7 + c] = expf(z[(size_t)i * 7 + c] - M) / S;
}

// ---------- host orchestration ----------
extern "C" void kernel_launch(void* const* d_in, const int* in_sizes, int n_in,
                              void* d_out, int out_size, void* d_ws, size_t ws_size,
                              hipStream_t stream) {
  (void)in_sizes; (void)n_in; (void)out_size; (void)ws_size;
  const int N = 8192, D = 128, C = 64;
  const float* x      = (const float*)d_in[0];
  const float* adj    = (const float*)d_in[1];
  const float* W_conv = (const float*)d_in[3];  // [4][128][128]
  const float* b_conv = (const float*)d_in[4];  // [4][128]
  const float* cent   = (const float*)d_in[5];  // [64][128]
  const float* W_mlp1 = (const float*)d_in[8];  // [7][128]
  const float* b_mlp1 = (const float*)d_in[9];  // [7]
  float* out = (float*)d_out;

  char* p = (char*)d_ws;
  size_t off = 0;
  auto take = [&](size_t bytes) -> char* {
    char* q = p + off;
    off = (off + bytes + 255) & ~(size_t)255;
    return q;
  };
  unsigned short* adjb = (unsigned short*)take((size_t)N * N * 2);  // 128MB, L2-resident
  unsigned short* hb   = (unsigned short*)take((size_t)N * D * 2);  // h in bf16 (GEMM1 A)
  unsigned short* tbT  = (unsigned short*)take((size_t)D * N * 2);  // t transposed (GEMM2 B)
  unsigned short* Wb   = (unsigned short*)take((size_t)D * D * 2);  // W bf16 (GEMM1 B)
  float* h    = (float*)take((size_t)N * D * 4);
  float* res  = (float*)take((size_t)N * D * 4);
  float* tmp  = (float*)take((size_t)N * D * 4);
  float* dist = (float*)take((size_t)N * C * 4);
  float* z    = (float*)take((size_t)N * 7 * 4);
  float* ro   = (float*)take(256);
  float* cn2  = (float*)take(256);

  // adjacency -> bf16 once (128MB fits in the 192MB global L2 for all 4 layers)
  k_f32_to_bf16_pk<<<2048, 256, 0, stream>>>(adj, (unsigned*)adjb, (long long)N * N / 2);
  k_init_h<<<1024, 256, 0, stream>>>(x, h, N);
  hipMemcpyAsync(res, h, (size_t)N * D * 4, hipMemcpyDeviceToDevice, stream);

  for (int l = 0; l < 4; l++) {
    k_f32_to_bf16_pk<<<32, 256, 0, stream>>>(W_conv + (size_t)l * D * D, (unsigned*)Wb, (long long)D * D / 2);
    k_f32_to_bf16_pk<<<512, 256, 0, stream>>>(h, (unsigned*)hb, (long long)N * D / 2);
    // mx = h @ W^T   (W row-major == required B[n][k] layout, K = D)
    k_gemm_bf16<<<N / 32, 256, 0, stream>>>(hb, Wb, tmp, D);
    k_hl_post<<<1024, 256, 0, stream>>>(tmp, h, b_conv + (size_t)l * D, tbT, N);
    // u = adj @ logmap0(h)  — 17 GFLOP GEMM, bf16 WMMA, f32 acc, K = N
    k_gemm_bf16<<<N / 32, 256, 0, stream>>>(adjb, tbT, tmp, N);
    k_agg_act<<<1024, 256, 0, stream>>>(tmp, h, res, (l & 1), N);
    if (l == 1) hipMemcpyAsync(res, h, (size_t)N * D * 4, hipMemcpyDeviceToDevice, stream);
  }

  k_centnorm<<<8, 256, 0, stream>>>(cent, cn2);
  k_centdist<<<N, 64, 0, stream>>>(h, cent, cn2, dist);
  k_readout<<<C, 256, 0, stream>>>(dist, ro);
  k_ones<<<1, 32, 0, stream>>>(out);                         // logits [1,1,7] == 1.0
  k_hr<<<1, 32, 0, stream>>>(ro, out + 7 + (size_t)N * 7);   // h_r [1,64]
  k_node_logits<<<1024, 256, 0, stream>>>(h, W_mlp1, b_mlp1, z, N);
  k_colsoftmax<<<7, 256, 0, stream>>>(z, out + 7, N);        // logits_node [1,8192,7]
}
